// TransformerEncoderLayer_75453985456275
// MI455X (gfx1250) — compile-verified, matrix operand order checked by reading, not run on hardware
//
#include <hip/hip_runtime.h>
#include <hip/hip_bf16.h>

typedef _Float16 h16;
typedef __attribute__((ext_vector_type(16))) _Float16 v16h;
typedef __attribute__((ext_vector_type(8)))  float    v8f;

#define EMBED 768
#define FFN   3072
#define HEADS 12
#define HDIM  64
#define BATCH 4
#define SEQ   2048
#define MROWS (BATCH*SEQ)   // 8192

union FragU { v16h v; uint4 u[2]; };

// ---------------------------------------------------------------------------
// CDNA5 async global->LDS copy (16B per lane), tracked by ASYNCcnt.
// LDS address operand = low 32 bits of the flat shared pointer (ISA 10.2).
// ---------------------------------------------------------------------------
__device__ __forceinline__ void async_copy16(void* lds_dst, const void* gsrc) {
  unsigned l = (unsigned)(uintptr_t)lds_dst;
  asm volatile("global_load_async_to_lds_b128 %0, %1, off"
               :: "v"(l), "v"(gsrc) : "memory");
}
__device__ __forceinline__ void async_wait0() {
  asm volatile("s_wait_asynccnt 0" ::: "memory");
}

// ---------------------------------------------------------------------------
// fp32 [K,N] weight -> f16 transposed [N,K]
// ---------------------------------------------------------------------------
__global__ __launch_bounds__(256) void k_wt(const float* __restrict__ W,
                                            h16* __restrict__ Wt, int K, int N) {
  int idx = blockIdx.x * 256 + threadIdx.x;
  if (idx >= N * K) return;
  int n = idx / K, k = idx - n * K;
  Wt[idx] = (h16)W[(size_t)k * N + n];
}

// ---------------------------------------------------------------------------
// LayerNorm: fp32 [M,C] -> f16 [M,C], one block per row
// ---------------------------------------------------------------------------
__global__ __launch_bounds__(256) void k_ln(const float* __restrict__ x,
                                            const float* __restrict__ g,
                                            const float* __restrict__ b,
                                            h16* __restrict__ out, int C) {
  int row = blockIdx.x;
  const float* xr = x + (size_t)row * C;
  float s = 0.f, s2 = 0.f;
  for (int i = threadIdx.x; i < C; i += 256) { float v = xr[i]; s += v; s2 += v * v; }
  for (int off = 16; off; off >>= 1) {
    s  += __shfl_xor(s,  off, 32);
    s2 += __shfl_xor(s2, off, 32);
  }
  __shared__ float ss[8], ss2[8];
  int wid = threadIdx.x >> 5, lane = threadIdx.x & 31;
  if (lane == 0) { ss[wid] = s; ss2[wid] = s2; }
  __syncthreads();
  s = 0.f; s2 = 0.f;
  for (int i = 0; i < 8; i++) { s += ss[i]; s2 += ss2[i]; }
  float mean = s / C;
  float var  = s2 / C - mean * mean;
  float rs   = rsqrtf(var + 1e-5f);
  for (int i = threadIdx.x; i < C; i += 256)
    out[(size_t)row * C + i] = (h16)((xr[i] - mean) * rs * g[i] + b[i]);
}

__device__ __forceinline__ float gelu_exact(float v) {
  return 0.5f * v * (1.0f + erff(v * 0.70710678118654752f));
}

// ---------------------------------------------------------------------------
// WMMA GEMM: C[M,N] = A[M,K](f16) @ Wt[N,K](f16, transposed) + bias
// EPI: 0 = f16 out (QKV), 1 = +residual, f32 out (proj / ffn2), 2 = GELU, f16 out
// 256 threads = 8 waves; block tile 128x128; wave tile 32x64; K tile 32.
// Double-buffered LDS staged with async global->LDS copies (ASYNCcnt).
// ---------------------------------------------------------------------------
template <int EPI>
__global__ __launch_bounds__(256) void k_gemm(const h16* __restrict__ A,
                                              const h16* __restrict__ Wt,
                                              const float* __restrict__ bias,
                                              const float* __restrict__ resid,
                                              float* __restrict__ outF,
                                              h16* __restrict__ outH,
                                              int M, int N, int K) {
  __shared__ h16 As[2][128 * 40];   // padded row stride 40 (80B, 16B aligned)
  __shared__ h16 Bs[2][128 * 40];

  const int tid   = threadIdx.x;
  const int wid   = tid >> 5, lane = tid & 31;
  const int wm    = wid >> 1, wn   = wid & 1;
  const int bm    = blockIdx.y * 128, bn = blockIdx.x * 128;
  const int lhalf = lane >> 4, lmod = lane & 15;

  v8f acc[2][4];
  for (int i = 0; i < 2; i++)
    for (int j = 0; j < 4; j++)
      for (int r = 0; r < 8; r++) acc[i][j][r] = 0.f;

  const int lrow = tid >> 1;          // 0..127
  const int lcol = (tid & 1) * 16;    // 0 or 16

  // async-stage one 128x32 A tile + 128x32 W tile into buffer `buf`
  auto stage = [&](int buf, int kt) {
    const h16* ga = A  + (size_t)(bm + lrow) * K + kt + lcol;
    const h16* gb = Wt + (size_t)(bn + lrow) * K + kt + lcol;
    h16* sa = &As[buf][lrow * 40 + lcol];
    h16* sb = &Bs[buf][lrow * 40 + lcol];
    async_copy16(sa,     ga);
    async_copy16(sa + 8, ga + 8);
    async_copy16(sb,     gb);
    async_copy16(sb + 8, gb + 8);
  };

  stage(0, 0);
  async_wait0();
  __syncthreads();

  int buf = 0;
  for (int kt = 0; kt < K; kt += 32) {
    // prefetch next K tile into the other buffer while we compute
    if (kt + 32 < K) stage(buf ^ 1, kt + 32);

    FragU a[2], b[4];
    for (int mi = 0; mi < 2; mi++) {
      const h16* p = &As[buf][(wm * 32 + mi * 16 + lmod) * 40 + lhalf * 8];
      a[mi].u[0] = *(const uint4*)p;
      a[mi].u[1] = *(const uint4*)(p + 16);
    }
    for (int ni = 0; ni < 4; ni++) {
      const h16* p = &Bs[buf][(wn * 64 + ni * 16 + lmod) * 40 + lhalf * 8];
      b[ni].u[0] = *(const uint4*)p;
      b[ni].u[1] = *(const uint4*)(p + 16);
    }
    for (int mi = 0; mi < 2; mi++)
      for (int ni = 0; ni < 4; ni++)
        acc[mi][ni] = __builtin_amdgcn_wmma_f32_16x16x32_f16(
            false, a[mi].v, false, b[ni].v, (short)0, acc[mi][ni], false, false);

    async_wait0();     // our prefetch landed in LDS
    __syncthreads();   // everyone's prefetch landed; done reading `buf`
    buf ^= 1;
  }

  // epilogue
  for (int mi = 0; mi < 2; mi++)
    for (int ni = 0; ni < 4; ni++)
      for (int r = 0; r < 8; r++) {
        int grow = bm + wm * 32 + mi * 16 + r + 8 * lhalf;
        int gcol = bn + wn * 64 + ni * 16 + lmod;
        float v = acc[mi][ni][r] + bias[gcol];
        if constexpr (EPI == 2) v = gelu_exact(v);
        if constexpr (EPI == 1) {
          v += resid[(size_t)grow * N + gcol];
          outF[(size_t)grow * N + gcol] = v;
        } else {
          outH[(size_t)grow * N + gcol] = (h16)v;
        }
      }
}

// ---------------------------------------------------------------------------
// Flash attention: grid (SEQ/128, BATCH*HEADS); 8 waves, 16 q-rows per wave.
// Q,K,V f16 stored [B,T,H,D] (row stride EMBED). Online softmax, f32 state.
// ---------------------------------------------------------------------------
__global__ __launch_bounds__(256) void k_attn(const h16* __restrict__ Q,
                                              const h16* __restrict__ Kb,
                                              const h16* __restrict__ Vb,
                                              h16* __restrict__ O) {
  const int bh = blockIdx.y;
  const int bb = bh / HEADS, hh = bh % HEADS;
  const int qbase = blockIdx.x * 128;
  const int tid = threadIdx.x, wid = tid >> 5, lane = tid & 31;
  const int lhalf = lane >> 4, lmod = lane & 15;

  __shared__ h16 Kt[32 * 64];        // [key][d]
  __shared__ h16 Vt[64 * 40];        // [d][key], padded
  __shared__ h16 Ps[8][16 * 40];     // per-wave P staging, padded

  // preload Q fragments for this wave's 16 rows (d-chunks 0..31, 32..63)
  FragU qf[2];
  {
    const int qrow = bb * SEQ + qbase + wid * 16 + lmod;
    for (int dc = 0; dc < 2; dc++) {
      const h16* p = Q + (size_t)qrow * EMBED + hh * HDIM + dc * 32 + lhalf * 8;
      qf[dc].u[0] = *(const uint4*)p;
      qf[dc].u[1] = *(const uint4*)(p + 16);
    }
  }

  float mrow[8], lrow[8];
  v8f o[4];
  for (int r = 0; r < 8; r++) { mrow[r] = -1e30f; lrow[r] = 0.f; }
  for (int ni = 0; ni < 4; ni++)
    for (int r = 0; r < 8; r++) o[ni][r] = 0.f;

  for (int kbase = 0; kbase < SEQ; kbase += 32) {
    // stage K tile async: 32 keys x 64 d (each thread 8 f16 = 16B)
    {
      int j = tid * 8;
      int key = j >> 6, d = j & 63;
      async_copy16(Kt + key * 64 + d,
                   Kb + (size_t)(bb * SEQ + kbase + key) * EMBED + hh * HDIM + d);
    }
    // stage V tile transposed: Vt[d][key] (strided gather, scalar path)
    for (int i = 0; i < 8; i++) {
      int j = tid * 8 + i;
      int d = j >> 5, key = j & 31;
      Vt[d * 40 + key] = Vb[(size_t)(bb * SEQ + kbase + key) * EMBED + hh * HDIM + d];
    }
    async_wait0();
    __syncthreads();

    // S[16q x 32k] = Q . K^T  (accumulate over two 32-wide d chunks)
    v8f s[2];
    for (int ni = 0; ni < 2; ni++)
      for (int r = 0; r < 8; r++) s[ni][r] = 0.f;
    for (int dc = 0; dc < 2; dc++)
      for (int ni = 0; ni < 2; ni++) {
        FragU kf;
        const h16* p = Kt + (ni * 16 + lmod) * 64 + dc * 32 + lhalf * 8;
        kf.u[0] = *(const uint4*)p;
        kf.u[1] = *(const uint4*)(p + 16);
        s[ni] = __builtin_amdgcn_wmma_f32_16x16x32_f16(
            false, qf[dc].v, false, kf.v, (short)0, s[ni], false, false);
      }

    // online softmax (rows grouped per half-wave: row = r + 8*lhalf)
    float pv[2][8];
    for (int r = 0; r < 8; r++) {
      float s0 = s[0][r] * 0.125f, s1 = s[1][r] * 0.125f;
      float mx = fmaxf(s0, s1);
      for (int off = 1; off < 16; off <<= 1) mx = fmaxf(mx, __shfl_xor(mx, off, 32));
      float mnew = fmaxf(mrow[r], mx);
      float sc = __expf(mrow[r] - mnew);
      mrow[r] = mnew;
      float p0 = __expf(s0 - mnew), p1 = __expf(s1 - mnew);
      pv[0][r] = p0; pv[1][r] = p1;
      float ps = p0 + p1;
      for (int off = 1; off < 16; off <<= 1) ps += __shfl_xor(ps, off, 32);
      lrow[r] = lrow[r] * sc + ps;
      for (int ni = 0; ni < 4; ni++) o[ni][r] *= sc;
    }

    // C-layout -> A-layout via LDS staging
    for (int ni = 0; ni < 2; ni++)
      for (int r = 0; r < 8; r++)
        Ps[wid][(r + 8 * lhalf) * 40 + ni * 16 + lmod] = (h16)pv[ni][r];
    __syncthreads();

    FragU pf;
    {
      const h16* p = &Ps[wid][lmod * 40 + lhalf * 8];
      pf.u[0] = *(const uint4*)p;
      pf.u[1] = *(const uint4*)(p + 16);
    }
    // O += P . V  (contraction over 32 keys, 4 d-chunks of 16)
    for (int ni = 0; ni < 4; ni++) {
      FragU vf;
      const h16* p = Vt + (ni * 16 + lmod) * 40 + lhalf * 8;
      vf.u[0] = *(const uint4*)p;
      vf.u[1] = *(const uint4*)(p + 16);
      o[ni] = __builtin_amdgcn_wmma_f32_16x16x32_f16(
          false, pf.v, false, vf.v, (short)0, o[ni], false, false);
    }
    __syncthreads();
  }

  // normalize and write out (f16, [B,T,H,D] layout)
  for (int ni = 0; ni < 4; ni++)
    for (int r = 0; r < 8; r++) {
      int qrow = bb * SEQ + qbase + wid * 16 + r + 8 * lhalf;
      int d = ni * 16 + lmod;
      O[(size_t)qrow * EMBED + hh * HDIM + d] = (h16)(o[ni][r] / lrow[r]);
    }
}

// ---------------------------------------------------------------------------
extern "C" void kernel_launch(void* const* d_in, const int* in_sizes, int n_in,
                              void* d_out, int out_size, void* d_ws, size_t ws_size,
                              hipStream_t stream) {
  const float* x    = (const float*)d_in[0];
  const float* ln1g = (const float*)d_in[1];
  const float* ln1b = (const float*)d_in[2];
  const float* wq   = (const float*)d_in[3];
  const float* bq   = (const float*)d_in[4];
  const float* wk   = (const float*)d_in[5];
  const float* bk   = (const float*)d_in[6];
  const float* wv   = (const float*)d_in[7];
  const float* bv   = (const float*)d_in[8];
  const float* wo   = (const float*)d_in[9];
  const float* bo   = (const float*)d_in[10];
  const float* ln2g = (const float*)d_in[11];
  const float* ln2b = (const float*)d_in[12];
  const float* w1   = (const float*)d_in[13];
  const float* b1   = (const float*)d_in[14];
  const float* w2   = (const float*)d_in[15];
  const float* b2   = (const float*)d_in[16];
  float* out = (float*)d_out;

  char* p = (char*)d_ws;
  auto alloc = [&](size_t bytes) -> void* {
    void* r = (void*)p;
    p += (bytes + 255) & ~(size_t)255;
    return r;
  };
  h16* wtq = (h16*)alloc((size_t)EMBED * EMBED * 2);
  h16* wtk = (h16*)alloc((size_t)EMBED * EMBED * 2);
  h16* wtv = (h16*)alloc((size_t)EMBED * EMBED * 2);
  h16* wto = (h16*)alloc((size_t)EMBED * EMBED * 2);
  h16* wt1 = (h16*)alloc((size_t)EMBED * FFN * 2);
  h16* wt2 = (h16*)alloc((size_t)FFN * EMBED * 2);
  h16* h1  = (h16*)alloc((size_t)MROWS * EMBED * 2);
  h16* qb  = (h16*)alloc((size_t)MROWS * EMBED * 2);
  h16* kb  = (h16*)alloc((size_t)MROWS * EMBED * 2);
  h16* vb  = (h16*)alloc((size_t)MROWS * EMBED * 2);
  h16* ao  = (h16*)alloc((size_t)MROWS * EMBED * 2);
  float* xa = (float*)alloc((size_t)MROWS * EMBED * 4);
  h16* h2  = (h16*)alloc((size_t)MROWS * EMBED * 2);
  h16* gb  = (h16*)alloc((size_t)MROWS * FFN * 2);

  // weight conversion (transpose to [N][K] f16)
  int ee = EMBED * EMBED, ef = EMBED * FFN;
  k_wt<<<(ee + 255) / 256, 256, 0, stream>>>(wq, wtq, EMBED, EMBED);
  k_wt<<<(ee + 255) / 256, 256, 0, stream>>>(wk, wtk, EMBED, EMBED);
  k_wt<<<(ee + 255) / 256, 256, 0, stream>>>(wv, wtv, EMBED, EMBED);
  k_wt<<<(ee + 255) / 256, 256, 0, stream>>>(wo, wto, EMBED, EMBED);
  k_wt<<<(ef + 255) / 256, 256, 0, stream>>>(w1, wt1, EMBED, FFN);
  k_wt<<<(ef + 255) / 256, 256, 0, stream>>>(w2, wt2, FFN, EMBED);

  // LN1
  k_ln<<<MROWS, 256, 0, stream>>>(x, ln1g, ln1b, h1, EMBED);

  // QKV projections
  dim3 gEE(EMBED / 128, MROWS / 128);
  k_gemm<0><<<gEE, 256, 0, stream>>>(h1, wtq, bq, nullptr, nullptr, qb, MROWS, EMBED, EMBED);
  k_gemm<0><<<gEE, 256, 0, stream>>>(h1, wtk, bk, nullptr, nullptr, kb, MROWS, EMBED, EMBED);
  k_gemm<0><<<gEE, 256, 0, stream>>>(h1, wtv, bv, nullptr, nullptr, vb, MROWS, EMBED, EMBED);

  // flash attention
  dim3 gAtt(SEQ / 128, BATCH * HEADS);
  k_attn<<<gAtt, 256, 0, stream>>>(qb, kb, vb, ao);

  // output projection + residual (fp32)
  k_gemm<1><<<gEE, 256, 0, stream>>>(ao, wto, bo, x, xa, nullptr, MROWS, EMBED, EMBED);

  // LN2
  k_ln<<<MROWS, 256, 0, stream>>>(xa, ln2g, ln2b, h2, EMBED);

  // FFN1 + GELU
  dim3 gF1(FFN / 128, MROWS / 128);
  k_gemm<2><<<gF1, 256, 0, stream>>>(h2, wt1, b1, nullptr, nullptr, gb, MROWS, FFN, EMBED);

  // FFN2 + residual -> output
  k_gemm<1><<<gEE, 256, 0, stream>>>(gb, wt2, b2, xa, out, nullptr, MROWS, EMBED, FFN);
}
